// AtnConv_76149770158306
// MI455X (gfx1250) — matrix operand order repaired and checked by reading, not compiled, not run on hardware
//
#include <hip/hip_runtime.h>
#include <hip/hip_bf16.h>
#include <stddef.h>
#include <stdint.h>

// ---------------------------------------------------------------------------
// Shapes (fixed by the reference): N=2, C=128, x2 48x48, x1 96x96.
// ---------------------------------------------------------------------------
#define HH 48
#define WW 48
#define NP (HH * WW)     // 2304 patches == spatial positions
#define CIN 128
#define KP (CIN * 9)     // 1152  (im2col K for 3x3 patches of x2)
#define NA (CIN * 16)    // 2048  (im2col N for 4x4 patches of x1)
#define H1 96
#define W1 96

typedef __attribute__((ext_vector_type(16))) _Float16       v16h;
typedef __attribute__((ext_vector_type(8)))  _Float16       v8h;
typedef __attribute__((ext_vector_type(16))) __bf16         v16bf;
typedef __attribute__((ext_vector_type(8)))  __bf16         v8bf;
typedef __attribute__((ext_vector_type(8)))  float          v8f;

union U16H  { v16h  v; v8h  h[2]; };
union U16BF { v16bf v; v8bf h[2]; };

static __device__ __forceinline__ unsigned short f2bf(float x) {
  unsigned int u = __float_as_uint(x);
  u += 0x7FFFu + ((u >> 16) & 1u);   // round-to-nearest-even
  return (unsigned short)(u >> 16);
}

// ---------------------------------------------------------------------------
// CDNA5 async global->LDS copy (ASYNCcnt-tracked), per CDNA5 ISA
// §15.18.3 / 08_async_tensor §4. LDS destination address = low 32 bits of the
// shared-memory pointer (hardware uses addr[31:0] as the LDS offset).
// ---------------------------------------------------------------------------
static __device__ __forceinline__ void async_b128(void* ldsdst, const void* gsrc) {
  unsigned l = (unsigned)(uintptr_t)ldsdst;
  unsigned long long g = (unsigned long long)(uintptr_t)gsrc;
  asm volatile("global_load_async_to_lds_b128 %0, %1, off"
               :: "v"(l), "v"(g) : "memory");
}
#define S_WAIT_ASYNC(N) asm volatile("s_wait_asynccnt %0" :: "n"(N) : "memory")

// ---------------------------------------------------------------------------
// im2col of x2 (3x3, stride 1, pad 1) -> f16 P[NP][KP]; per-patch L2 norms;
// hole mask mm[p] = (patch of `mask` sums to 0).
// ---------------------------------------------------------------------------
__global__ __launch_bounds__(128) void build_p(const float* __restrict__ x2n,
                                               const float* __restrict__ maskn,
                                               _Float16* __restrict__ P,
                                               float* __restrict__ norms,
                                               float* __restrict__ mm) {
  __shared__ float red[128];
  const int p = blockIdx.x;
  const int pi = p / WW, pj = p % WW;
  const int c = threadIdx.x;
  const float* xc = x2n + (size_t)c * (HH * WW);
  float ss = 0.f;
#pragma unroll
  for (int t = 0; t < 9; ++t) {
    const int dy = t / 3, dx = t % 3;
    const int yy = pi + dy - 1, xx = pj + dx - 1;
    float val = 0.f;
    if (yy >= 0 && yy < HH && xx >= 0 && xx < WW) val = xc[yy * WW + xx];
    P[(size_t)p * KP + c * 9 + t] = (_Float16)val;
    ss += val * val;
  }
  red[c] = ss;
  __syncthreads();
  for (int s = 64; s > 0; s >>= 1) {
    if (c < s) red[c] += red[c + s];
    __syncthreads();
  }
  if (c == 0) {
    norms[p] = sqrtf(red[0]);
    float msum = 0.f;
#pragma unroll
    for (int t = 0; t < 9; ++t) {
      const int dy = t / 3, dx = t % 3;
      const int yy = pi + dy - 1, xx = pj + dx - 1;
      if (yy >= 0 && yy < HH && xx >= 0 && xx < WW) msum += maskn[yy * WW + xx];
    }
    mm[p] = (msum == 0.f) ? 1.f : 0.f;
  }
}

// ---------------------------------------------------------------------------
// im2col of x1 (4x4, stride 2, pad 1), stored TRANSPOSED: AT[m][p], m = c*16 +
// ky*4 + kx. This makes the transfer GEMM a pure NT GEMM (contiguous K for
// both operands -> 16B async-to-LDS staging, no LDS transpose).
// ---------------------------------------------------------------------------
__global__ __launch_bounds__(128) void build_a(const float* __restrict__ x1n,
                                               unsigned short* __restrict__ AT) {
  const int p = blockIdx.x;
  const int pi = p / WW, pj = p % WW;
  const int c = threadIdx.x;
  const float* xc = x1n + (size_t)c * (H1 * W1);
#pragma unroll
  for (int t = 0; t < 16; ++t) {
    const int ky = t >> 2, kx = t & 3;
    const int yy = 2 * pi + ky - 1, xx = 2 * pj + kx - 1;
    float val = 0.f;
    if (yy >= 0 && yy < H1 && xx >= 0 && xx < W1) val = xc[yy * W1 + xx];
    AT[(size_t)(c * 16 + t) * NP + p] = f2bf(val);
  }
}

// ---------------------------------------------------------------------------
// NT GEMM, f16: C[m][n] = sum_k A[m,k]*B[n,k]; A,B row-major (ld = K).
// 256 threads = 8 waves; block tile 64x64; wave tile 16x32 (one A frag feeds
// two WMMAs); K step 32; double-buffered LDS filled by async-to-LDS copies.
// Fragment k order is contiguous: lane half h holds k = 16h..16h+15 for A and B.
// ---------------------------------------------------------------------------
__global__ __launch_bounds__(256) void gemm_nt_f16(const _Float16* __restrict__ A,
                                                   const _Float16* __restrict__ B,
                                                   float* __restrict__ C,
                                                   int N, int K) {
  __shared__ __align__(32) _Float16 At[2][64][32];
  __shared__ __align__(32) _Float16 Bt[2][64][32];
  const int tid = threadIdx.x;
  const int mBase = blockIdx.y * 64;
  const int nBase = blockIdx.x * 64;
  const int wave = tid >> 5, lane = tid & 31;
  const int mT = (wave & 3) * 16;
  const int nS = (wave >> 2) * 32;
  const int l15 = lane & 15;
  const bool hiHalf = lane >= 16;
  const int srow = tid >> 2, sch = (tid & 3) * 8;   // 64 rows x 4 x 16B chunks

  v8f acc0 = {}, acc1 = {};
  const int nk = K >> 5;

  // Prologue: stage K-tile 0 into buffer 0 (2 async b128 per thread).
  async_b128(&At[0][srow][sch], &A[(size_t)(mBase + srow) * K + sch]);
  async_b128(&Bt[0][srow][sch], &B[(size_t)(nBase + srow) * K + sch]);

  for (int kt = 0; kt < nk; ++kt) {
    const int cur = kt & 1;
    if (kt + 1 < nk) {
      const int k0 = (kt + 1) << 5;
      async_b128(&At[cur ^ 1][srow][sch], &A[(size_t)(mBase + srow) * K + k0 + sch]);
      async_b128(&Bt[cur ^ 1][srow][sch], &B[(size_t)(nBase + srow) * K + k0 + sch]);
      S_WAIT_ASYNC(2);   // drain previous stage; keep the 2 just-issued in flight
    } else {
      S_WAIT_ASYNC(0);
    }
    __syncthreads();

    U16H a, b0, b1;
    const int alo = hiHalf ? 16 : 0;
    const int blo = hiHalf ? 16 : 0;
    a.h[0]  = *(const v8h*)&At[cur][mT + l15][alo];        // K = alo..alo+7
    a.h[1]  = *(const v8h*)&At[cur][mT + l15][alo + 8];    // K = alo+8..alo+15
    b0.h[0] = *(const v8h*)&Bt[cur][nS + l15][blo];
    b0.h[1] = *(const v8h*)&Bt[cur][nS + l15][blo + 8];
    b1.h[0] = *(const v8h*)&Bt[cur][nS + 16 + l15][blo];
    b1.h[1] = *(const v8h*)&Bt[cur][nS + 16 + l15][blo + 8];

    acc0 = __builtin_amdgcn_wmma_f32_16x16x32_f16(false, a.v, false, b0.v,
                                                  (short)0, acc0, false, false);
    acc1 = __builtin_amdgcn_wmma_f32_16x16x32_f16(false, a.v, false, b1.v,
                                                  (short)0, acc1, false, false);
    __syncthreads();   // all reads done before this buffer is refilled
  }

  const int mg = mBase + mT + (hiHalf ? 8 : 0);
  const int ng = nBase + nS + l15;
#pragma unroll
  for (int r = 0; r < 8; ++r) {
    C[(size_t)(mg + r) * N + ng]      = acc0[r];
    C[(size_t)(mg + r) * N + ng + 16] = acc1[r];
  }
}

// ---------------------------------------------------------------------------
// NT GEMM, bf16 (same structure, v_wmma_f32_16x16x32_bf16).
// ---------------------------------------------------------------------------
__global__ __launch_bounds__(256) void gemm_nt_bf16(const unsigned short* __restrict__ A,
                                                    const unsigned short* __restrict__ B,
                                                    float* __restrict__ C,
                                                    int N, int K) {
  __shared__ __align__(32) unsigned short At[2][64][32];
  __shared__ __align__(32) unsigned short Bt[2][64][32];
  const int tid = threadIdx.x;
  const int mBase = blockIdx.y * 64;
  const int nBase = blockIdx.x * 64;
  const int wave = tid >> 5, lane = tid & 31;
  const int mT = (wave & 3) * 16;
  const int nS = (wave >> 2) * 32;
  const int l15 = lane & 15;
  const bool hiHalf = lane >= 16;
  const int srow = tid >> 2, sch = (tid & 3) * 8;

  v8f acc0 = {}, acc1 = {};
  const int nk = K >> 5;

  async_b128(&At[0][srow][sch], &A[(size_t)(mBase + srow) * K + sch]);
  async_b128(&Bt[0][srow][sch], &B[(size_t)(nBase + srow) * K + sch]);

  for (int kt = 0; kt < nk; ++kt) {
    const int cur = kt & 1;
    if (kt + 1 < nk) {
      const int k0 = (kt + 1) << 5;
      async_b128(&At[cur ^ 1][srow][sch], &A[(size_t)(mBase + srow) * K + k0 + sch]);
      async_b128(&Bt[cur ^ 1][srow][sch], &B[(size_t)(nBase + srow) * K + k0 + sch]);
      S_WAIT_ASYNC(2);
    } else {
      S_WAIT_ASYNC(0);
    }
    __syncthreads();

    U16BF a, b0, b1;
    const int alo = hiHalf ? 16 : 0;
    const int blo = hiHalf ? 16 : 0;
    a.h[0]  = *(const v8bf*)&At[cur][mT + l15][alo];
    a.h[1]  = *(const v8bf*)&At[cur][mT + l15][alo + 8];
    b0.h[0] = *(const v8bf*)&Bt[cur][nS + l15][blo];
    b0.h[1] = *(const v8bf*)&Bt[cur][nS + l15][blo + 8];
    b1.h[0] = *(const v8bf*)&Bt[cur][nS + 16 + l15][blo];
    b1.h[1] = *(const v8bf*)&Bt[cur][nS + 16 + l15][blo + 8];

    acc0 = __builtin_amdgcn_wmma_f32_16x16x32_bf16(false, a.v, false, b0.v,
                                                   (short)0, acc0, false, false);
    acc1 = __builtin_amdgcn_wmma_f32_16x16x32_bf16(false, a.v, false, b1.v,
                                                   (short)0, acc1, false, false);
    __syncthreads();
  }

  const int mg = mBase + mT + (hiHalf ? 8 : 0);
  const int ng = nBase + nS + l15;
#pragma unroll
  for (int r = 0; r < 8; ++r) {
    C[(size_t)(mg + r) * N + ng]      = acc0[r];
    C[(size_t)(mg + r) * N + ng + 16] = acc1[r];
  }
}

// ---------------------------------------------------------------------------
// Per-position softmax over 2304 patch scores + masking + 4-neighbor 1.5x
// boost (analytic mask_c) + clamp -> bf16. One block per position s.
// ---------------------------------------------------------------------------
__global__ __launch_bounds__(256) void softmax_boost(const float* __restrict__ ST,
                                                     const float* __restrict__ norms,
                                                     const float* __restrict__ mm,
                                                     const float* __restrict__ mask_all_n,
                                                     unsigned short* __restrict__ Yt) {
  __shared__ float red[256];
  const int s = blockIdx.x;
  const int si = s / WW, sj = s % WW;
  const int tid = threadIdx.x;
  const float ma = mask_all_n[s];
  float v[9];
  float mx = -1e30f;
#pragma unroll
  for (int e = 0; e < 9; ++e) {
    const int p = e * 256 + tid;
    const float raw = ST[(size_t)s * NP + p];
    const float val = raw / fmaxf(norms[p], 1e-4f) * mm[p] * ma * 10.0f;
    v[e] = val;
    mx = fmaxf(mx, val);
  }
  red[tid] = mx;
  __syncthreads();
  for (int t = 128; t > 0; t >>= 1) {
    if (tid < t) red[tid] = fmaxf(red[tid], red[tid + t]);
    __syncthreads();
  }
  const float gmax = red[0];
  __syncthreads();
  float sum = 0.f;
#pragma unroll
  for (int e = 0; e < 9; ++e) {
    v[e] = __expf(v[e] - gmax);
    sum += v[e];
  }
  red[tid] = sum;
  __syncthreads();
  for (int t = 128; t > 0; t >>= 1) {
    if (tid < t) red[tid] += red[tid + t];
    __syncthreads();
  }
  const float inv = 1.f / red[0];
#pragma unroll
  for (int e = 0; e < 9; ++e) {
    const int p = e * 256 + tid;
    const int pi = p / WW, pj = p % WW;
    float yv = v[e] * inv;
    const int man = abs(pi - si) + abs(pj - sj);
    if (man == 1) yv *= 1.5f;               // yi + yi*mask_c*0.5
    yv *= mm[p] * ma;
    yv = fmaxf(yv, 1e-8f);
    Yt[(size_t)s * NP + p] = f2bf(yv);
  }
}

// ---------------------------------------------------------------------------
// col2im for the stride-2 transposed conv: each output pixel gathers its
// 2x2 covering (patch, tap) contributions from T[NP][NA], /4.
// ---------------------------------------------------------------------------
__global__ __launch_bounds__(256) void col2im(const float* __restrict__ T,
                                              float* __restrict__ yout) {
  const int idx = blockIdx.x * 256 + threadIdx.x;
  if (idx >= CIN * H1 * W1) return;
  const int v = idx % W1;
  const int u = (idx / W1) % H1;
  const int o = idx / (H1 * W1);
  float acc = 0.f;
  for (int kh = (u & 1); kh < 4; kh += 2) {
    const int a = u + kh - 2;
    if (a < 0 || a > 2 * HH - 2) continue;
    const int i = a >> 1;
    for (int kw = (v & 1); kw < 4; kw += 2) {
      const int b = v + kw - 2;
      if (b < 0 || b > 2 * WW - 2) continue;
      const int j = b >> 1;
      acc += T[(size_t)(i * WW + j) * NA + o * 16 + (3 - kh) * 4 + (3 - kw)];
    }
  }
  yout[idx] = acc * 0.25f;
}

// ---------------------------------------------------------------------------
// Fused 4-group dilated 3x3 conv (dil 1/2/4/8), 128 -> 16 ch per group,
// bias + ReLU, groups concatenated. ~2.7 GFLOP: direct VALU.
// ---------------------------------------------------------------------------
__global__ __launch_bounds__(256) void fuse_conv(const float* __restrict__ y,
                                                 const float* __restrict__ fw,
                                                 const float* __restrict__ fb,
                                                 float* __restrict__ out) {
  const int idx = blockIdx.x * 256 + threadIdx.x;
  if (idx >= 2 * 64 * H1 * W1) return;
  const int v = idx % W1;
  const int u = (idx / W1) % H1;
  const int ch = (idx / (H1 * W1)) % 64;
  const int n = idx / (64 * H1 * W1);
  const int g = ch >> 4;
  const int d = 1 << g;
  const float* yb = y + (size_t)n * CIN * H1 * W1;
  const float* w = fw + (size_t)ch * CIN * 9;
  float acc = fb[ch];
  for (int c = 0; c < CIN; ++c) {
    const float* yc = yb + (size_t)c * H1 * W1;
    const float* wc = w + c * 9;
#pragma unroll
    for (int ky = 0; ky < 3; ++ky) {
      const int yy = u + (ky - 1) * d;
      if (yy < 0 || yy >= H1) continue;
#pragma unroll
      for (int kx = 0; kx < 3; ++kx) {
        const int xx = v + (kx - 1) * d;
        if (xx < 0 || xx >= W1) continue;
        acc += yc[yy * W1 + xx] * wc[ky * 3 + kx];
      }
    }
  }
  out[idx] = fmaxf(acc, 0.f);
}

// ---------------------------------------------------------------------------
// Workspace layout (per-batch buffers reused via stream serialization).
// Total ~71.5 MB; everything L2-resident (192 MB).
// ---------------------------------------------------------------------------
static constexpr size_t P_OFF   = 0;
static constexpr size_t P_BY    = (size_t)NP * KP * 2;          // 5,308,416
static constexpr size_t A_OFF   = P_OFF + P_BY;
static constexpr size_t A_BY    = (size_t)NP * NA * 2;          // 9,437,184
static constexpr size_t ST_OFF  = A_OFF + A_BY;
static constexpr size_t ST_BY   = (size_t)NP * NP * 4;          // 21,233,664
static constexpr size_t YT_OFF  = ST_OFF + ST_BY;
static constexpr size_t YT_BY   = (size_t)NP * NP * 2;          // 10,616,832
static constexpr size_t T_OFF   = YT_OFF + YT_BY;
static constexpr size_t T_BY    = (size_t)NP * NA * 4;          // 18,874,368
static constexpr size_t NRM_OFF = T_OFF + T_BY;
static constexpr size_t NRM_BY  = (size_t)NP * 4;
static constexpr size_t MM_OFF  = NRM_OFF + NRM_BY;
static constexpr size_t MM_BY   = (size_t)NP * 4;
static constexpr size_t Y_OFF   = MM_OFF + MM_BY;

extern "C" void kernel_launch(void* const* d_in, const int* in_sizes, int n_in,
                              void* d_out, int out_size, void* d_ws, size_t ws_size,
                              hipStream_t stream) {
  (void)in_sizes; (void)n_in; (void)out_size; (void)ws_size;
  const float* x1       = (const float*)d_in[0];   // (2,128,96,96)
  const float* x2       = (const float*)d_in[1];   // (2,128,48,48)
  const float* mask     = (const float*)d_in[2];   // (2,1,48,48)
  const float* mask_all = (const float*)d_in[3];   // (2,1,48,48)
  const float* fuse_w   = (const float*)d_in[4];   // (4,16,128,3,3)
  const float* fuse_b   = (const float*)d_in[5];   // (4,16)

  char* ws = (char*)d_ws;
  _Float16*       P     = (_Float16*)(ws + P_OFF);
  unsigned short* AT    = (unsigned short*)(ws + A_OFF);
  float*          ST    = (float*)(ws + ST_OFF);
  unsigned short* Yt    = (unsigned short*)(ws + YT_OFF);
  float*          T     = (float*)(ws + T_OFF);
  float*          norms = (float*)(ws + NRM_OFF);
  float*          mmv   = (float*)(ws + MM_OFF);
  float*          y     = (float*)(ws + Y_OFF);

  for (int n = 0; n < 2; ++n) {
    build_p<<<NP, 128, 0, stream>>>(x2 + (size_t)n * CIN * HH * WW,
                                    mask + (size_t)n * HH * WW, P, norms, mmv);
    build_a<<<NP, 128, 0, stream>>>(x1 + (size_t)n * CIN * H1 * W1, AT);

    dim3 g1(NP / 64, NP / 64);                       // 36 x 36
    gemm_nt_f16<<<g1, 256, 0, stream>>>(P, P, ST, NP, KP);

    softmax_boost<<<NP, 256, 0, stream>>>(ST, norms, mmv,
                                          mask_all + (size_t)n * HH * WW, Yt);

    dim3 g2(NA / 64, NP / 64);                       // 32 x 36
    gemm_nt_bf16<<<g2, 256, 0, stream>>>(Yt, AT, T, NA, NP);

    col2im<<<(CIN * H1 * W1 + 255) / 256, 256, 0, stream>>>(
        T, y + (size_t)n * CIN * H1 * W1);
  }

  fuse_conv<<<(2 * 64 * H1 * W1 + 255) / 256, 256, 0, stream>>>(
      y, fuse_w, fuse_b, (float*)d_out);
}
